// Network_10488310137541
// MI455X (gfx1250) — matrix operand order repaired and checked by reading, not compiled, not run on hardware
//
#include <hip/hip_runtime.h>
#include <hip/hip_bf16.h>

// ---------------------------------------------------------------------------
// Binarized MLP: out = sigmoid( sign(x @ sign(W1)^T) @ sign(W2)^T )
// B=16384, D_IN=D_H=4096, D_OUT=1
// Core: bf16 WMMA GEMM (v_wmma_f32_16x16x32_bf16) fed by double-buffered
// async global->LDS staging (global_load_async_to_lds_b128 / s_wait_asynccnt),
// batched ds_load fragment fetch (partial dscnt waits), global_prefetch warmup,
// fused sign+reduce epilogue folding layer 2 into per-row atomics.
// ---------------------------------------------------------------------------

typedef __attribute__((ext_vector_type(16))) __bf16 v16bf;
typedef __attribute__((ext_vector_type(8)))  float  v8f;

union Frag {
    int4  q[2];   // two 16B chunks per lane
    v16bf v;      // 16 bf16 elements -> 8 VGPRs
};

#define B_ROWS 16384
#define D_DIM  4096
#define KSTAGE 64                 // K elements staged per LDS buffer
#define LDS_STRIDE 72             // 64 + 8 pad elements; 144 B = 9*16 B rows
#define BUF_ELEMS (128 * LDS_STRIDE)
#define BUF_BYTES (BUF_ELEMS * 2)

// --- helpers ---------------------------------------------------------------

__device__ __forceinline__ unsigned short f32_to_bf16_rne(float f) {
    unsigned int u = __float_as_uint(f);
    u += 0x7FFFu + ((u >> 16) & 1u);       // round-to-nearest-even
    return (unsigned short)(u >> 16);
}

__device__ __forceinline__ unsigned int pack_bf16x2(float lo, float hi) {
    return (unsigned int)f32_to_bf16_rne(lo) |
           ((unsigned int)f32_to_bf16_rne(hi) << 16);
}

// Async 16B global->LDS copy (ASYNCcnt-tracked). Inline asm: portable across
// ROCm 7.2 and amdgpu-toolchain (the clang builtin arity differs between them).
__device__ __forceinline__ void async_cp16(unsigned lds_byte_addr,
                                           const void* gaddr) {
    asm volatile("global_load_async_to_lds_b128 %0, %1, off"
                 :: "v"(lds_byte_addr), "v"((unsigned long long)gaddr)
                 : "memory");
}

__device__ __forceinline__ void wait_asynccnt0() {
    asm volatile("s_wait_asynccnt 0" ::: "memory");
}

// --- prep kernels ----------------------------------------------------------

// x (f32) -> bf16, 8 elements / thread
__global__ __launch_bounds__(256) void k_prep_x(const float* __restrict__ x,
                                                unsigned short* __restrict__ xb) {
    long i = (long)(blockIdx.x * 256 + threadIdx.x) * 8;
    float4 a = *(const float4*)(x + i);
    float4 b = *(const float4*)(x + i + 4);
    uint4 o;
    o.x = pack_bf16x2(a.x, a.y);
    o.y = pack_bf16x2(a.z, a.w);
    o.z = pack_bf16x2(b.x, b.y);
    o.w = pack_bf16x2(b.z, b.w);
    *(uint4*)(xb + i) = o;
}

// W1 (f32) -> sign(W1) as bf16 +/-1.0, 8 elements / thread
__global__ __launch_bounds__(256) void k_prep_w1(const float* __restrict__ w,
                                                 unsigned short* __restrict__ wb) {
    long i = (long)(blockIdx.x * 256 + threadIdx.x) * 8;
    float4 a = *(const float4*)(w + i);
    float4 b = *(const float4*)(w + i + 4);
    const unsigned int P = 0x3F80u, N = 0xBF80u;   // bf16 +1.0 / -1.0
    uint4 o;
    o.x = (a.x >= 0.f ? P : N) | ((a.y >= 0.f ? P : N) << 16);
    o.y = (a.z >= 0.f ? P : N) | ((a.w >= 0.f ? P : N) << 16);
    o.z = (b.x >= 0.f ? P : N) | ((b.y >= 0.f ? P : N) << 16);
    o.w = (b.z >= 0.f ? P : N) | ((b.w >= 0.f ? P : N) << 16);
    *(uint4*)(wb + i) = o;
}

// W2 (f32, 4096) -> sign as f32 +/-1.0
__global__ __launch_bounds__(256) void k_prep_w2(const float* __restrict__ w2,
                                                 float* __restrict__ s2) {
    int i = blockIdx.x * 256 + threadIdx.x;
    s2[i] = (w2[i] >= 0.f) ? 1.f : -1.f;
}

// --- main GEMM: async-LDS double buffer + WMMA + fused epilogue ------------
// Grid: (N/128, M/128) = (32, 128); 256 threads = 8 wave32.
// Wave (wm,wn) in 2x4 arrangement owns a 64(M) x 32(N) region:
//   4 M-tiles x 2 N-tiles of 16x16 f32 accumulators; K staged 64 at a time.
__global__ __launch_bounds__(256) void k_gemm(const unsigned short* __restrict__ xb,
                                              const unsigned short* __restrict__ w1b,
                                              const float* __restrict__ s2,
                                              float* __restrict__ accum) {
    __shared__ __align__(16) unsigned short shA[2][BUF_ELEMS];
    __shared__ __align__(16) unsigned short shB[2][BUF_ELEMS];

    const int K = D_DIM;
    const int tid    = threadIdx.x;
    const int wave   = tid >> 5;
    const int lane   = tid & 31;
    const int lane_r = lane & 15;
    const int hi     = lane >> 4;

    const int wm = wave & 1;         // 2 M sub-bands of 64
    const int wn = wave >> 1;        // 4 N sub-bands of 32
    const int MbaseWG = blockIdx.y * 128;
    const int NbaseWG = blockIdx.x * 128;

    // ---- async copy assignment: 1024 16B chunks per 128x64 tile,
    //      4 chunks per thread: rows (tid>>3)+32*i, column chunk tid&7.
    const int cp_row0 = tid >> 3;
    const int cp_cole = (tid & 7) * 8;                 // element column
    const unsigned ldsA0 = (unsigned)(unsigned long long)(&shA[0][0]);
    const unsigned ldsB0 = (unsigned)(unsigned long long)(&shB[0][0]);

    const unsigned short* gArow[4];
    const unsigned short* gBrow[4];
    unsigned lAoff[4], lBoff[4];
#pragma unroll
    for (int i = 0; i < 4; ++i) {
        int r = cp_row0 + 32 * i;
        gArow[i] = xb  + (long)(MbaseWG + r) * K + cp_cole;
        gBrow[i] = w1b + (long)(NbaseWG + r) * K + cp_cole;
        lAoff[i] = (unsigned)((r * LDS_STRIDE + cp_cole) * 2);
        lBoff[i] = (unsigned)((r * LDS_STRIDE + cp_cole) * 2);
    }

    // ---- fragment read geometry (contiguous k = 16*hi + s, same order for A and B) ----
    // A 16x32: lanes 0-15 -> K 0..15 contiguous, lanes 16-31 -> K 16..31
    const int aoff = hi * 16;
    // B 32x16: lanes 0-15 -> K 0..15 contiguous, lanes 16-31 -> K 16..31
    const int boff = hi * 16;

    v8f acc[4][2];
    const v8f vzero = {0.f, 0.f, 0.f, 0.f, 0.f, 0.f, 0.f, 0.f};
#pragma unroll
    for (int mt = 0; mt < 4; ++mt)
#pragma unroll
        for (int nt = 0; nt < 2; ++nt)
            acc[mt][nt] = vzero;

    // prologue: stage k0 = 0 into buffer 0
#pragma unroll
    for (int i = 0; i < 4; ++i) {
        async_cp16(ldsA0 + lAoff[i], gArow[i]);
        async_cp16(ldsB0 + lBoff[i], gBrow[i]);
    }

    int buf = 0;
    for (int k0 = 0; k0 < K; k0 += KSTAGE) {
        // make buffer `buf` (issued last stage / prologue) visible to all waves
        wait_asynccnt0();
        __syncthreads();

        // overlap: stage next K block into the other buffer
        if (k0 + KSTAGE < K) {
            const unsigned dst = (buf ^ 1) ? (unsigned)BUF_BYTES : 0u;
            const int ge = k0 + KSTAGE;
#pragma unroll
            for (int i = 0; i < 4; ++i) {
                async_cp16(ldsA0 + dst + lAoff[i], gArow[i] + ge);
                async_cp16(ldsB0 + dst + lBoff[i], gBrow[i] + ge);
            }
        }
        // warm GL2 for stage+2 (global_prefetch_b8)
        if (k0 + 2 * KSTAGE < K) {
            __builtin_prefetch(gArow[0] + k0 + 2 * KSTAGE, 0, 1);
            __builtin_prefetch(gBrow[0] + k0 + 2 * KSTAGE, 0, 1);
        }

        const unsigned short* As = &shA[buf][0];
        const unsigned short* Bs = &shB[buf][0];

        // Batch ALL fragment loads for this stage (24 ds_load_b128, in-order
        // DS returns -> partial s_wait_dscnt), then run the 16 WMMAs.
        Frag a[2][4], b[2][2];
#pragma unroll
        for (int kh = 0; kh < 2; ++kh) {
            const int kk = kh * 32;
#pragma unroll
            for (int nt = 0; nt < 2; ++nt) {
                int base = (wn * 32 + nt * 16 + lane_r) * LDS_STRIDE + kk + boff;
                b[kh][nt].q[0] = *(const int4*)(Bs + base);
                b[kh][nt].q[1] = *(const int4*)(Bs + base + 8);
            }
#pragma unroll
            for (int mt = 0; mt < 4; ++mt) {
                int base = (wm * 64 + mt * 16 + lane_r) * LDS_STRIDE + kk + aoff;
                a[kh][mt].q[0] = *(const int4*)(As + base);
                a[kh][mt].q[1] = *(const int4*)(As + base + 8);
            }
        }
        // keep the scheduler from sinking the loads back between the WMMAs
        __builtin_amdgcn_sched_barrier(0);
#pragma unroll
        for (int kh = 0; kh < 2; ++kh)
#pragma unroll
            for (int mt = 0; mt < 4; ++mt)
#pragma unroll
                for (int nt = 0; nt < 2; ++nt)
                    acc[mt][nt] = __builtin_amdgcn_wmma_f32_16x16x32_bf16(
                        false, a[kh][mt].v, false, b[kh][nt].v,
                        (short)0, acc[mt][nt], false, false);
        buf ^= 1;
    }

    // Epilogue: C/D layout -> element r holds row M = r + 8*hi, col N = lane_r.
    // Fold layer 2: per-row sum of sign(h)*sign(W2) over this wave's 32 cols.
    const int Nbase = NbaseWG + wn * 32;
    const int Mbase = MbaseWG + wm * 64;
    float s2v[2];
#pragma unroll
    for (int nt = 0; nt < 2; ++nt)
        s2v[nt] = s2[Nbase + nt * 16 + lane_r];

#pragma unroll
    for (int mt = 0; mt < 4; ++mt) {
#pragma unroll
        for (int r = 0; r < 8; ++r) {
            float contrib = 0.f;
#pragma unroll
            for (int nt = 0; nt < 2; ++nt) {
                float v = acc[mt][nt][r];
                contrib += (v >= 0.f) ? s2v[nt] : -s2v[nt];
            }
            // reduce across the 16 lanes of this half (rows differ per half)
            contrib += __shfl_xor(contrib, 8, 32);
            contrib += __shfl_xor(contrib, 4, 32);
            contrib += __shfl_xor(contrib, 2, 32);
            contrib += __shfl_xor(contrib, 1, 32);
            if (lane_r == 0) {
                atomicAdd(&accum[Mbase + mt * 16 + hi * 8 + r], contrib);
            }
        }
    }
}

// --- finalize: sigmoid + threshold -----------------------------------------
__global__ __launch_bounds__(256) void k_finalize(const float* __restrict__ accum,
                                                  float* __restrict__ out) {
    int b = blockIdx.x * 256 + threadIdx.x;
    float v = accum[b];
    float o = 1.f / (1.f + __expf(-v));
    out[b] = o;                               // sigmoid output [B,1]
    out[B_ROWS + b] = (v >= 0.f) ? 1.f : 0.f; // y_hat (sigmoid(v)>=0.5 <=> v>=0)
}

// ---------------------------------------------------------------------------

extern "C" void kernel_launch(void* const* d_in, const int* in_sizes, int n_in,
                              void* d_out, int out_size, void* d_ws, size_t ws_size,
                              hipStream_t stream) {
    const float* x  = (const float*)d_in[0];   // [16384, 4096]
    const float* W1 = (const float*)d_in[1];   // [4096, 4096]
    const float* W2 = (const float*)d_in[2];   // [1, 4096]
    float* out = (float*)d_out;                // 32768 floats (out ++ y_hat)

    char* ws = (char*)d_ws;
    const size_t XB_BYTES  = (size_t)B_ROWS * D_DIM * 2;   // 134,217,728
    const size_t W1B_BYTES = (size_t)D_DIM * D_DIM * 2;    //  33,554,432
    const size_t S2_BYTES  = (size_t)D_DIM * 4;            //      16,384
    unsigned short* xb   = (unsigned short*)ws;
    unsigned short* w1b  = (unsigned short*)(ws + XB_BYTES);
    float*          s2   = (float*)(ws + XB_BYTES + W1B_BYTES);
    float*          accm = (float*)(ws + XB_BYTES + W1B_BYTES + S2_BYTES);

    hipMemsetAsync(accm, 0, (size_t)B_ROWS * sizeof(float), stream);

    k_prep_x <<<(B_ROWS * (long)D_DIM) / (8 * 256), 256, 0, stream>>>(x, xb);
    k_prep_w1<<<((long)D_DIM * D_DIM) / (8 * 256), 256, 0, stream>>>(W1, w1b);
    k_prep_w2<<<D_DIM / 256, 256, 0, stream>>>(W2, s2);

    dim3 grid(D_DIM / 128, B_ROWS / 128);   // (32, 128)
    k_gemm<<<grid, 256, 0, stream>>>(xb, w1b, s2, accm);

    k_finalize<<<B_ROWS / 256, 256, 0, stream>>>(accm, out);
}
